// GaussianRenderer_55027120996693
// MI455X (gfx1250) — matrix operand order repaired and checked by reading, not compiled, not run on hardware
//
#include <hip/hip_runtime.h>
#include <hip/hip_bf16.h>

typedef __attribute__((ext_vector_type(16))) _Float16 v16h;
typedef __attribute__((ext_vector_type(8)))  _Float16 v8h;
typedef __attribute__((ext_vector_type(8)))  float    v8f;

#define Bn 2
#define Nn 512
#define Cn 32
#define Hn 200
#define Wn 200
#define HWn (Hn * Wn)
#define NTILES ((HWn + 127) / 128)   // 313 pixel tiles of 128 per batch

__launch_bounds__(256)
__global__ void gs_render_kernel(const float* __restrict__ feat,   // (B,N,C)
                                 const float* __restrict__ means,  // (B,N,3)
                                 const float* __restrict__ cov,    // (B,N,6)
                                 const float* __restrict__ opac,   // (B,N,1)
                                 float* __restrict__ out)          // (B,C,H,W)
{
    __shared__ __align__(16) _Float16 featT[Cn * Nn];   // 32 KB: featT[c*Nn + n]
    __shared__ __align__(16) float    gp[Nn][8];        // 16 KB: projected params

    const int tid  = threadIdx.x;
    const int b    = blockIdx.x / NTILES;
    const int tile = blockIdx.x % NTILES;

    // ---- stage features transposed, f32 -> f16 (coalesced global reads) ----
    for (int i = tid; i < Nn * Cn; i += 256) {
        const int n = i >> 5;
        const int c = i & 31;
        featT[c * Nn + n] = (_Float16)feat[((size_t)b * Nn + n) * Cn + c];
    }

    // ---- stage per-Gaussian 2D projection params ----
    for (int n = tid; n < Nn; n += 256) {
        const size_t gi = (size_t)b * Nn + n;
        const float m0 = means[gi * 3 + 0];
        const float m1 = means[gi * 3 + 1];
        const float s_xx = cov[gi * 6 + 0];
        const float s_xy = cov[gi * 6 + 1];
        const float s_yy = cov[gi * 6 + 3];
        float o = opac[gi];
        o = (o > 0.05f) ? o : 0.0f;

        const float mu_u = 100.0f - 100.0f * m1;   // H/2 - SH*y
        const float mu_v = 100.0f - 100.0f * m0;   // W/2 - SW*x
        const float Cuu  = 10000.0f * s_yy + 0.3f;
        const float Cvv  = 10000.0f * s_xx + 0.3f;
        const float Cuv  = 10000.0f * s_xy;
        const float det  = Cuu * Cvv - Cuv * Cuv;
        const bool valid = det > 0.0f;
        const float inv_det = valid ? 1.0f / det : 0.0f;
        if (!valid) o = 0.0f;

        gp[n][0] = mu_u;
        gp[n][1] = mu_v;
        gp[n][2] = Cvv * inv_det;    // Iuu
        gp[n][3] = Cuu * inv_det;    // Ivv
        gp[n][4] = -Cuv * inv_det;   // Iuv
        gp[n][5] = o;
        gp[n][6] = 0.0f;
        gp[n][7] = 0.0f;
    }
    __syncthreads();

    // ---- per-wave: 16-pixel strip, 32 channels via two 16x16 WMMA tiles ----
    const int lane = tid & 31;
    const int half = lane >> 4;        // 0 or 1
    const int pix  = lane & 15;        // pixel within strip / row within A tile
    const int wave = tid >> 5;
    const int p    = tile * 128 + wave * 16 + pix;
    const int pc   = (p < HWn) ? p : (HWn - 1);    // clamp: keep EXEC all-ones
    const float fu = (float)(pc / Wn);
    const float fv = (float)(pc % Wn);

    v8f acc0 = {};     // channels 0..15
    v8f acc1 = {};     // channels 16..31
    float T = 1.0f;    // running transmittance for this lane's pixel

    for (int kb = 0; kb < 16; ++kb) {
        // sequential compositing scan over this chunk of 32 Gaussians
        float warr[32];
        #pragma unroll
        for (int e = 0; e < 32; ++e) {
            const int n = kb * 32 + e;
            const float4 g0 = *(const float4*)&gp[n][0];   // mu_u, mu_v, Iuu, Ivv
            const float2 g1 = *(const float2*)&gp[n][4];   // Iuv, opac
            const float du = fu - g0.x;
            const float dv = fv - g0.y;
            const float power = -0.5f * (g0.z * du * du + g0.w * dv * dv)
                                - g1.x * du * dv;
            float a = g1.y * __expf(fminf(power, 0.0f));
            a = fminf(a, 0.99f);
            a = (power > 0.0f || a < (1.0f / 255.0f)) ? 0.0f : a;
            warr[e] = T * a;
            T *= (1.0f - a);
        }

        // pack B fragment: lanes 0-15 carry K=0..15, lanes 16-31 carry K=16..31
        v16h bfrag;
        #pragma unroll
        for (int e = 0; e < 16; ++e) {
            const float wl = warr[e];
            const float wh = warr[e + 16];
            bfrag[e] = (_Float16)(half ? wh : wl);   // constant-indexed, stays in VGPRs
        }

        // A fragments from LDS (row M = lane&15 = channel; K split by lane half:
        // elems 0..15 -> K = half*16 .. +15)
        const int kbase = kb * 32 + half * 16;
        const v8h a0lo = *(const v8h*)&featT[pix * Nn + kbase];
        const v8h a0hi = *(const v8h*)&featT[pix * Nn + kbase + 8];
        const v8h a1lo = *(const v8h*)&featT[(16 + pix) * Nn + kbase];
        const v8h a1hi = *(const v8h*)&featT[(16 + pix) * Nn + kbase + 8];
        v16h afrag0, afrag1;
        #pragma unroll
        for (int i = 0; i < 8; ++i) {
            afrag0[i]     = a0lo[i];
            afrag0[i + 8] = a0hi[i];
            afrag1[i]     = a1lo[i];
            afrag1[i + 8] = a1hi[i];
        }

        acc0 = __builtin_amdgcn_wmma_f32_16x16x32_f16(
            false, afrag0, false, bfrag, (short)0, acc0, false, false);
        acc1 = __builtin_amdgcn_wmma_f32_16x16x32_f16(
            false, afrag1, false, bfrag, (short)0, acc1, false, false);
    }

    // ---- store: C/D layout is VGPR r, M = r + 8*half, N = lane&15 = pixel ----
    if (p < HWn) {
        #pragma unroll
        for (int r = 0; r < 8; ++r) {
            const int ch = r + 8 * half;
            out[((size_t)b * Cn + ch) * HWn + p]        = acc0[r];
            out[((size_t)b * Cn + ch + 16) * HWn + p]   = acc1[r];
        }
    }
}

__launch_bounds__(256)
__global__ void gs_count_kernel(const float* __restrict__ opac, float* __restrict__ out_scalar)
{
    __shared__ float red[256];
    const int tid = threadIdx.x;
    float s = 0.0f;
    for (int i = tid; i < Bn * Nn; i += 256)
        s += (opac[i] > 0.05f) ? 1.0f : 0.0f;
    red[tid] = s;
    __syncthreads();
    for (int off = 128; off > 0; off >>= 1) {
        if (tid < off) red[tid] += red[tid + off];
        __syncthreads();
    }
    if (tid == 0) out_scalar[0] = red[0] / (float)Bn;
}

extern "C" void kernel_launch(void* const* d_in, const int* in_sizes, int n_in,
                              void* d_out, int out_size, void* d_ws, size_t ws_size,
                              hipStream_t stream)
{
    const float* feat  = (const float*)d_in[0];   // (B,N,C)
    const float* means = (const float*)d_in[1];   // (B,N,3)
    const float* cov   = (const float*)d_in[2];   // (B,N,6)
    const float* opac  = (const float*)d_in[3];   // (B,N,1)
    float* out = (float*)d_out;                   // bev (B,C,H,W) flat, then scalar

    gs_render_kernel<<<Bn * NTILES, 256, 0, stream>>>(feat, means, cov, opac, out);
    gs_count_kernel<<<1, 256, 0, stream>>>(opac, out + (size_t)Bn * Cn * HWn);
}
